// StructureModel_18846316494849
// MI455X (gfx1250) — matrix operand order repaired and checked by reading, not compiled, not run on hardware
//
#include <hip/hip_runtime.h>
#include <hip/hip_bf16.h>

#define NNODE 10000
#define NEDGE 640000
#define EMBED 256
#define EDGED 64

typedef __attribute__((ext_vector_type(16))) __bf16 v16bf;
typedef __attribute__((ext_vector_type(8)))  __bf16 v8bf;
typedef __attribute__((ext_vector_type(8)))  float  v8f;
typedef unsigned __attribute__((ext_vector_type(4))) u32x4;
typedef int      __attribute__((ext_vector_type(8))) i32x8;
typedef int      __attribute__((ext_vector_type(4))) i32x4;

// LDS pitch for the staged B tile: 128 bf16 = 256B row + 4B pad (TDM pad_interval=5)
#define BPITCH 260

// ---------- helpers ----------
__device__ __forceinline__ __bf16 f2bf(float f) {
    unsigned u = __float_as_uint(f);
    unsigned r = (u + 0x7FFFu + ((u >> 16) & 1u)) >> 16;   // round-to-nearest-even
    unsigned short s = (unsigned short)r;
    __bf16 b;
    __builtin_memcpy(&b, &s, 2);
    return b;
}

// order-preserving encoding of float for unsigned atomicMax
__device__ __forceinline__ unsigned ordf(float f) {
    unsigned u = __float_as_uint(f);
    return (u & 0x80000000u) ? ~u : (u | 0x80000000u);
}
__device__ __forceinline__ float deordf(unsigned u) {
    return __uint_as_float((u & 0x80000000u) ? (u ^ 0x80000000u) : ~u);
}

// ---------- embedding gather: h0[i,c] = emb[x[i], c] ----------
__global__ void k_gather(const int* __restrict__ x, const float* __restrict__ emb,
                         float* __restrict__ h) {
    long t = (long)blockIdx.x * blockDim.x + threadIdx.x;
    if (t >= (long)NNODE * EMBED) return;
    int node = (int)(t / EMBED), c = (int)(t % EMBED);
    h[t] = emb[(long)x[node] * EMBED + c];
}

// ---------- f32 -> bf16 ----------
__global__ void k_tobf16(const float* __restrict__ src, __bf16* __restrict__ dst, long n) {
    long t = (long)blockIdx.x * blockDim.x + threadIdx.x;
    if (t < n) dst[t] = f2bf(src[t]);
}

// ---------- layer GEMM: C[M x Nn] f32 = A[M x K] bf16 * B[K x Nn] bf16 (both row-major) ----
// One wave per 16x16 C tile; 8 waves per block.
__global__ void k_gemm_bf16(const __bf16* __restrict__ A, const __bf16* __restrict__ B,
                            float* __restrict__ C, int M, int Nn, int K) {
    const int wave = threadIdx.x >> 5;
    const int lane = threadIdx.x & 31;
    const int tilesN = Nn >> 4;
    const long totalTiles = (long)(M >> 4) * tilesN;
    const long tileIdx = (long)blockIdx.x * 8 + wave;
    if (tileIdx >= totalTiles) return;            // wave-uniform: EXEC stays all-1s
    const int tm = (int)(tileIdx / tilesN);
    const int tn = (int)(tileIdx % tilesN);

    v8f acc = {};
    const int rowA  = tm * 16 + (lane & 15);
    const int kbase = (lane >> 4) * 8;                    // A half-select
    const int kB    = (lane & 15) + 16 * (lane >> 4);     // B lane's K

    for (int k0 = 0; k0 < K; k0 += 32) {
        // A tile 16x32: per lane K in [k0+kbase, +8) and [k0+kbase+16, +8)
        const __bf16* abase = A + (long)rowA * K + k0 + kbase;
        v8bf alo = *(const v8bf*)(abase);
        v8bf ahi = *(const v8bf*)(abase + 16);
        v16bf a;
#pragma unroll
        for (int i = 0; i < 8; ++i) { a[i] = alo[i]; a[8 + i] = ahi[i]; }

        // row-major K x Nn: lane column n = lane&15, K in [k0+kbase, +8) and [k0+kbase+16, +8)
        v16bf b;
#pragma unroll
        for (int i = 0; i < 8; ++i) {
            b[i]     = B[(long)(k0 + kbase + i) * Nn + tn * 16 + (lane & 15)];
            b[8 + i] = B[(long)(k0 + kbase + 16 + i) * Nn + tn * 16 + (lane & 15)];
        }
        if (k0 + 32 < K) __builtin_prefetch(abase + 32, 0, 0);

        acc = __builtin_amdgcn_wmma_f32_16x16x32_bf16(
            false, a, false, b, (short)0, acc, false, false);
    }

    // C layout: VGPR r -> row r + 8*(lane>>4), col lane&15
    const int cm = tm * 16 + 8 * (lane >> 4);
    const int cn = tn * 16 + (lane & 15);
#pragma unroll
    for (int r = 0; r < 8; ++r)
        C[(long)(cm + r) * Nn + cn] = acc[r];
}

// ---------- final GEMM: C = h @ h^T  (M=Nn=NNODE, K=128) ----------
// Block = 8 waves -> 128 (M) x 16 (N) of C. The 16 h-rows forming B for this
// column tile are staged into LDS once per block via the Tensor Data Mover
// (tensor_load_to_lds with +1-DWORD/row padding to spread LDS banks), then all
// waves read the transposed B operand from LDS (ds path) while A streams from
// global (L2-resident) with b128 loads.
__global__ void k_gemm_ht(const __bf16* __restrict__ A, float* __restrict__ C) {
    const int K  = 128;
    const int Nn = NNODE;
    __shared__ __align__(16) char smem[16 * BPITCH];

    const int wave = threadIdx.x >> 5;
    const int lane = threadIdx.x & 31;
    const int tn   = blockIdx.x;                 // column tile 0..624
    const int tm   = blockIdx.y * 8 + wave;      // row tile

    // ---- stage B tile: h[tn*16 .. tn*16+15][0..127] -> LDS (16 x 260B) ----
#if __has_builtin(__builtin_amdgcn_tensor_load_to_lds) && __has_builtin(__builtin_amdgcn_s_wait_tensorcnt)
    if (wave == 0) {
        unsigned long long gaddr =
            (unsigned long long)(A + (unsigned long long)tn * 16 * K);
        unsigned ldsOff = (unsigned)(unsigned long long)(&smem[0]);

        u32x4 g0;
        g0[0] = 1u;                                         // count=1, user mode
        g0[1] = ldsOff;                                     // lds_addr
        g0[2] = (unsigned)(gaddr & 0xFFFFFFFFull);          // global_addr[31:0]
        g0[3] = (unsigned)((gaddr >> 32) & 0x01FFFFFFull)   // global_addr[56:32]
                | (2u << 30);                               // type=2 (image)

        i32x8 g1;
        g1[0] = (int)((1u << 16)      // data_size = 1 (2 bytes)
                    | (1u << 20)      // pad_enable
                    | (5u << 22));    // pad_interval: 2^(5+1)=64 DWORDs = 256B
                                      // pad_amount = 0 -> 1 DWORD
        g1[1] = (int)(128u << 16);    // tensor_dim0[15:0] = 128
        g1[2] = (int)((NNODE & 0xFFFFu) << 16);  // dim0 hi=0 | tensor_dim1 lo
        g1[3] = (int)(128u << 16);    // tensor_dim1 hi=0 | tile_dim0 = 128
        g1[4] = 16;                   // tile_dim1 = 16, tile_dim2 = 0
        g1[5] = 128;                  // tensor_dim0_stride lo = 128
        g1[6] = 0;                    // stride hi | tensor_dim1_stride lo
        g1[7] = 0;

        i32x4 z4 = {0, 0, 0, 0};
        i32x8 z8 = {0, 0, 0, 0, 0, 0, 0, 0};
        __builtin_amdgcn_tensor_load_to_lds(g0, g1, z4, z4, z8, 0);
        __builtin_amdgcn_s_wait_tensorcnt(0);
    }
#else
    for (int t = threadIdx.x; t < 16 * K; t += blockDim.x) {
        int r = t >> 7, c = t & 127;
        *(__bf16*)(smem + r * BPITCH + c * 2) =
            A[((long)tn * 16 + r) * K + c];
    }
#endif
    __syncthreads();

    if (tm >= (NNODE >> 4)) return;              // wave-uniform exit after barrier

    v8f acc = {};
    const int rowA  = tm * 16 + (lane & 15);
    const int kbase = (lane >> 4) * 8;
    const int kB    = (lane & 15) + 16 * (lane >> 4);

#pragma unroll
    for (int k0 = 0; k0 < K; k0 += 32) {
        const __bf16* abase = A + (long)rowA * K + k0 + kbase;
        v8bf alo = *(const v8bf*)(abase);
        v8bf ahi = *(const v8bf*)(abase + 16);
        v16bf a;
#pragma unroll
        for (int i = 0; i < 8; ++i) { a[i] = alo[i]; a[8 + i] = ahi[i]; }

        // B[k][n] = h[tn*16+n][k] from padded LDS tile (row pitch 260B)
        v16bf b;
        const char* bcol = smem + (lane & 15) * BPITCH + (k0 + kbase) * 2;
#pragma unroll
        for (int i = 0; i < 8; ++i) {
            b[i]     = *(const __bf16*)(bcol + i * 2);
            b[8 + i] = *(const __bf16*)(bcol + (16 + i) * 2);
        }

        acc = __builtin_amdgcn_wmma_f32_16x16x32_bf16(
            false, a, false, b, (short)0, acc, false, false);
    }

    const int cm = tm * 16 + 8 * (lane >> 4);
    const int cn = tn * 16 + (lane & 15);
#pragma unroll
    for (int r = 0; r < 8; ++r)
        C[(long)(cm + r) * Nn + cn] = acc[r];
}

// ---------- qi = xt@q, kj = xt@k (thread per node) ----------
__global__ void k_qikj(const float* __restrict__ xt, const float* __restrict__ q,
                       const float* __restrict__ kvec, float* __restrict__ qi,
                       float* __restrict__ kj, int co) {
    int i = blockIdx.x * blockDim.x + threadIdx.x;
    if (i >= NNODE) return;
    float sq = 0.f, sk = 0.f;
    const float* row = xt + (long)i * co;
    for (int c = 0; c < co; ++c) { sq += row[c] * q[c]; sk += row[c] * kvec[c]; }
    qi[i] = sq; kj[i] = sk;
}

// ---------- we_e[d] = sum_c We[d,c] * e[c] ----------
__global__ void k_wee(const float* __restrict__ We, const float* __restrict__ e,
                      float* __restrict__ wee, int co) {
    int d = threadIdx.x;
    if (d >= EDGED) return;
    float s = 0.f;
    for (int c = 0; c < co; ++c) s += We[(long)d * co + c] * e[c];
    wee[d] = s;
}

// ---------- per-node init: m = ord(-inf), denom = 0 ----------
__global__ void k_initnode(unsigned* __restrict__ mo, float* __restrict__ denom) {
    int i = blockIdx.x * blockDim.x + threadIdx.x;
    if (i >= NNODE) return;
    mo[i] = ordf(-__builtin_inff());
    denom[i] = 0.f;
}
__global__ void k_zero(float* __restrict__ p, long n) {
    long t = (long)blockIdx.x * blockDim.x + threadIdx.x;
    if (t < n) p[t] = 0.f;
}

// ---------- alpha = lrelu(qi[dst]+kj[src]+ae, 0.2); segment max via atomicMax ----------
__global__ void k_alpha(const int* __restrict__ src, const int* __restrict__ dst,
                        const float* __restrict__ eattr, const float* __restrict__ wee,
                        const float* __restrict__ qi, const float* __restrict__ kj,
                        float* __restrict__ alpha, unsigned* __restrict__ mo) {
    int e = blockIdx.x * blockDim.x + threadIdx.x;
    if (e >= NEDGE) return;
    const float* row = eattr + (long)e * EDGED;
    float ae = 0.f;
#pragma unroll 8
    for (int d = 0; d < EDGED; ++d) ae += row[d] * wee[d];
    float v = qi[dst[e]] + kj[src[e]] + ae;
    v = v >= 0.f ? v : 0.2f * v;
    alpha[e] = v;
    atomicMax(&mo[dst[e]], ordf(v));
}

// ---------- ex = exp(alpha - m[dst]); denom += ex ----------
__global__ void k_ex(const int* __restrict__ dst, const float* __restrict__ alpha,
                     const unsigned* __restrict__ mo, float* __restrict__ ex,
                     float* __restrict__ denom) {
    int e = blockIdx.x * blockDim.x + threadIdx.x;
    if (e >= NEDGE) return;
    float v = __expf(alpha[e] - deordf(mo[dst[e]]));
    ex[e] = v;
    atomicAdd(&denom[dst[e]], v);
}

// ---------- agg[dst,c] += (ex/denom[dst]) * xt[src,c]; block per edge ----------
__global__ void k_scatter(const int* __restrict__ src, const int* __restrict__ dst,
                          const float* __restrict__ ex, const float* __restrict__ denom,
                          const float* __restrict__ xt, float* __restrict__ agg, int co) {
    int e = blockIdx.x;
    int c = threadIdx.x;
    int d = dst[e], s = src[e];
    float a = ex[e] / (denom[d] + 1e-16f);
    atomicAdd(&agg[(long)d * co + c], a * xt[(long)s * co + c]);
}

// ---------- h_next = (agg + b), optional lrelu 0.01 ----------
__global__ void k_finish(const float* __restrict__ agg, const float* __restrict__ b,
                         float* __restrict__ hout, int co, int do_lrelu) {
    long t = (long)blockIdx.x * blockDim.x + threadIdx.x;
    if (t >= (long)NNODE * co) return;
    int c = (int)(t % co);
    float v = agg[t] + b[c];
    if (do_lrelu) v = v >= 0.f ? v : 0.01f * v;
    hout[t] = v;
}

// =======================================================================
extern "C" void kernel_launch(void* const* d_in, const int* in_sizes, int n_in,
                              void* d_out, int out_size, void* d_ws, size_t ws_size,
                              hipStream_t stream) {
    static const int CI[5] = {256, 128, 64, 32, 64};
    static const int CO[5] = {128, 64, 32, 64, 128};

    const int*   x     = (const int*)d_in[0];
    const int*   eidx  = (const int*)d_in[1];   // [2, E]
    const float* eattr = (const float*)d_in[2]; // [E, 64]
    const float* emb   = (const float*)d_in[4]; // [10000, 256]
    const int*   src   = eidx;
    const int*   dst   = eidx + NEDGE;

    // params: 5 layers x leaves sorted {W, We, b, e, k, q}
    const float* W[5]; const float* We[5]; const float* bb[5];
    const float* ev[5]; const float* kv[5]; const float* qv[5];
    for (int l = 0; l < 5; ++l) {
        int base = 5 + 6 * l;
        W[l]  = (const float*)d_in[base + 0];
        We[l] = (const float*)d_in[base + 1];
        bb[l] = (const float*)d_in[base + 2];
        ev[l] = (const float*)d_in[base + 3];
        kv[l] = (const float*)d_in[base + 4];
        qv[l] = (const float*)d_in[base + 5];
    }

    // workspace carve
    char* p = (char*)d_ws;
    auto carve = [&](size_t bytes) -> void* {
        void* r = (void*)p;
        p += (bytes + 255) & ~(size_t)255;
        return r;
    };
    float*    hA    = (float*)carve((size_t)NNODE * 256 * 4);
    float*    hB    = (float*)carve((size_t)NNODE * 256 * 4);
    float*    xt    = (float*)carve((size_t)NNODE * 128 * 4);
    __bf16*   hb    = (__bf16*)carve((size_t)NNODE * 256 * 2);
    __bf16*   Wb    = (__bf16*)carve((size_t)256 * 128 * 2);
    float*    qi    = (float*)carve((size_t)NNODE * 4);
    float*    kj    = (float*)carve((size_t)NNODE * 4);
    float*    wee   = (float*)carve((size_t)EDGED * 4);
    float*    alpha = (float*)carve((size_t)NEDGE * 4);
    float*    exb   = (float*)carve((size_t)NEDGE * 4);
    unsigned* mo    = (unsigned*)carve((size_t)NNODE * 4);
    float*    denom = (float*)carve((size_t)NNODE * 4);
    float*    agg   = (float*)carve((size_t)NNODE * 128 * 4);

    const int TB = 256;
    auto cdiv = [](long a, long b) { return (int)((a + b - 1) / b); };

    // h0 = emb[x]
    k_gather<<<cdiv((long)NNODE * EMBED, TB), TB, 0, stream>>>(x, emb, hA);

    float* hin = hA;
    float* hout = hB;
    for (int l = 0; l < 5; ++l) {
        const int ci = CI[l], co = CO[l];

        // bf16 copies
        k_tobf16<<<cdiv((long)NNODE * ci, TB), TB, 0, stream>>>(hin, hb, (long)NNODE * ci);
        k_tobf16<<<cdiv((long)ci * co, TB), TB, 0, stream>>>(W[l], Wb, (long)ci * co);

        // xt = h @ W  (WMMA)
        long tiles = (long)(NNODE / 16) * (co / 16);
        k_gemm_bf16<<<cdiv(tiles, 8), 256, 0, stream>>>(hb, Wb, xt, NNODE, co, ci);

        // attention scalars
        k_qikj<<<cdiv(NNODE, TB), TB, 0, stream>>>(xt, qv[l], kv[l], qi, kj, co);
        k_wee<<<1, 64, 0, stream>>>(We[l], ev[l], wee, co);

        // init segment state
        k_initnode<<<cdiv(NNODE, TB), TB, 0, stream>>>(mo, denom);
        k_zero<<<cdiv((long)NNODE * co, TB), TB, 0, stream>>>(agg, (long)NNODE * co);

        // edge pipeline
        k_alpha<<<cdiv(NEDGE, TB), TB, 0, stream>>>(src, dst, eattr, wee, qi, kj, alpha, mo);
        k_ex<<<cdiv(NEDGE, TB), TB, 0, stream>>>(dst, alpha, mo, exb, denom);
        k_scatter<<<NEDGE, co, 0, stream>>>(src, dst, exb, denom, xt, agg, co);

        // bias (+lrelu except last layer)
        k_finish<<<cdiv((long)NNODE * co, TB), TB, 0, stream>>>(agg, bb[l], hout, co, l < 4 ? 1 : 0);

        float* t = hin; hin = hout; hout = t;
    }

    // final: out = h @ h^T, h is N x 128; TDM-staged B tiles
    k_tobf16<<<cdiv((long)NNODE * 128, TB), TB, 0, stream>>>(hin, hb, (long)NNODE * 128);
    dim3 grid(NNODE / 16, (NNODE / 16 + 7) / 8);   // 625 x 79
    k_gemm_ht<<<grid, 256, 0, stream>>>(hb, (float*)d_out);
}
